// DbrxExperts_64501818851514
// MI455X (gfx1250) — matrix operand order repaired and checked by reading, not compiled, not run on hardware
//
#include <hip/hip_runtime.h>

// Problem constants (from reference setup_inputs)
#define T_TOK  4096
#define DMODEL 1024
#define IFFN   2048
#define NEXP   16
#define TOPK   4

typedef __attribute__((ext_vector_type(16))) _Float16 v16h;
typedef __attribute__((ext_vector_type(8)))  _Float16 v8h;
typedef __attribute__((ext_vector_type(8)))  float    v8f;

// ---- WMMA fragment loaders (f16 source) ------------------------------------
// A-matrix (16x32 f16), per ISA 7.12.2: lane L holds row M = L%16.
// Lanes 0-15: K = kb+0..7 (elems 0-7), kb+16..23 (elems 8-15)
// Lanes 16-31: K = kb+8..15,          kb+24..31
__device__ __forceinline__ v16h load_a_f16(const _Float16* row, int kb, int khalf) {
  v8h lo = *reinterpret_cast<const v8h*>(row + kb + 8 * khalf);
  v8h hi = *reinterpret_cast<const v8h*>(row + kb + 16 + 8 * khalf);
  v16h a;
#pragma unroll
  for (int i = 0; i < 8; ++i) { a[i] = lo[i]; a[i + 8] = hi[i]; }
  return a;
}

// B-matrix (32x16 f16) for computing X @ W^T: lane L holds column N = L%16,
// which is row N of W (row-major, contiguous K). Lanes 0-15: K = kb+0..7, kb+16..23,
// lanes 16-31: K = kb+8..15, kb+24..31. Two global_load_b128, zero conversions.
__device__ __forceinline__ v16h load_b_f16row(const _Float16* __restrict__ row,
                                              int kb, int khalf) {
  const _Float16* p = row + kb + 8 * khalf;
  v8h lo = *reinterpret_cast<const v8h*>(p);
  v8h hi = *reinterpret_cast<const v8h*>(p + 16);
  v16h b;
#pragma unroll
  for (int i = 0; i < 8; ++i) { b[i] = lo[i]; b[i + 8] = hi[i]; }
  return b;
}

// fast silu using v_rcp_f32 (avoids IEEE div_scale/div_fixup chain)
__device__ __forceinline__ float fast_silu(float g) {
  return g * __builtin_amdgcn_rcpf(1.f + __expf(-g));
}

// ---- Kernel 0: one-time fp32 -> f16 bulk conversion (weights) ---------------
__global__ __launch_bounds__(256) void cvt_f32_f16_kernel(
    const float* __restrict__ src, _Float16* __restrict__ dst) {
  const size_t i = ((size_t)blockIdx.x * 256 + threadIdx.x) * 8;
  float4 a = *reinterpret_cast<const float4*>(src + i);
  float4 b = *reinterpret_cast<const float4*>(src + i + 4);
  v8h o;
  o[0] = (_Float16)a.x; o[1] = (_Float16)a.y; o[2] = (_Float16)a.z; o[3] = (_Float16)a.w;
  o[4] = (_Float16)b.x; o[5] = (_Float16)b.y; o[6] = (_Float16)b.z; o[7] = (_Float16)b.w;
  *reinterpret_cast<v8h*>(dst + i) = o;
}

// ---- Kernel 1: router (logits -> top-4 gates), expert list scatter, X->f16 --
__global__ __launch_bounds__(256) void moe_router_kernel(
    const float* __restrict__ hs, const float* __restrict__ rw,
    _Float16* __restrict__ xh, int* __restrict__ counts,
    int* __restrict__ etok, float* __restrict__ egate) {
  const int lane = threadIdx.x & 31;
  const int wave = threadIdx.x >> 5;
  const int t = blockIdx.x * 8 + wave;   // one token per wave, 8 waves/block

  float acc[NEXP];
#pragma unroll
  for (int e = 0; e < NEXP; ++e) acc[e] = 0.f;

  const float* xrow = hs + (size_t)t * DMODEL;
  for (int d = lane; d < DMODEL; d += 32) {
    float xv = xrow[d];
#pragma unroll
    for (int e = 0; e < NEXP; ++e) acc[e] += xv * rw[e * DMODEL + d];
  }
  // cross-lane reduce all 16 logits (every lane ends with full values)
#pragma unroll
  for (int e = 0; e < NEXP; ++e) {
#pragma unroll
    for (int off = 16; off >= 1; off >>= 1) acc[e] += __shfl_xor(acc[e], off, 32);
  }

  // top-4 of softmax == top-4 of logits; renormalized weights need only
  // exp(l_i - max) over the selected set. Ties pick lowest index (matches lax.top_k).
  float m = acc[0];
#pragma unroll
  for (int e = 1; e < NEXP; ++e) m = fmaxf(m, acc[e]);
  unsigned sel = 0u;
  int kidx[TOPK]; float kw[TOPK]; float s4 = 0.f;
#pragma unroll
  for (int k = 0; k < TOPK; ++k) {
    int bi = 0; float bv = -3.4e38f;
#pragma unroll
    for (int e = 0; e < NEXP; ++e)
      if (!(sel & (1u << e)) && acc[e] > bv) { bv = acc[e]; bi = e; }
    sel |= (1u << bi);
    kidx[k] = bi;
    float w = __expf(acc[bi] - m);
    kw[k] = w; s4 += w;
  }
  const float inv_s4 = __builtin_amdgcn_rcpf(s4);
  if (lane == 0) {
#pragma unroll
    for (int k = 0; k < TOPK; ++k) {
      int e = kidx[k];
      int pos = atomicAdd(&counts[e], 1);
      etok[e * T_TOK + pos]  = (t << 2) | k;   // pack token id + slot
      egate[e * T_TOK + pos] = kw[k] * inv_s4;
    }
  }

  // convert this block's 8 rows of hidden_states to f16 (read once, 8 MB total)
  const size_t base = (size_t)blockIdx.x * 8 * DMODEL;
  for (int i = threadIdx.x; i < 8 * DMODEL; i += 256)
    xh[base + i] = (_Float16)hs[base + i];
}

// ---- Kernel 2: fused SwiGLU expert FFN, 64-token tile per block -------------
// grid = NEXP * (T/64) blocks (worst case; empty tiles exit). 256 thr = 8 waves.
// Dynamic LDS: act tile 64 x 2048 f16 (256 KB) + token/gate staging.
// Each wave job processes TWO 16-row token tiles sharing the same B fragments,
// halving weight-fragment loads per WMMA.
__global__ __launch_bounds__(256) void moe_expert_kernel(
    const _Float16* __restrict__ xh,
    const _Float16* __restrict__ wsh,   // [E, 2I, D] f16
    const _Float16* __restrict__ w2h,   // [E, D, I]  f16
    const int* __restrict__ counts,
    const int* __restrict__ etok, const float* __restrict__ egate,
    float* __restrict__ yslot) {        // [T, 4, D] gate-scaled contributions
  extern __shared__ char smem_raw[];
  _Float16* actS = reinterpret_cast<_Float16*>(smem_raw);                 // 64*2048
  int*   stokS  = reinterpret_cast<int*>(smem_raw + (size_t)64 * IFFN * 2);
  float* sgateS = reinterpret_cast<float*>(smem_raw + (size_t)64 * IFFN * 2 + 64 * 4);

  const int e = blockIdx.x >> 6;       // 64 tiles per expert
  const int tileStart = (blockIdx.x & 63) * 64;
  const int cnt = counts[e];
  if (tileStart >= cnt) return;        // uniform across block (before barriers)

  const int lane = threadIdx.x & 31;
  const int wave = threadIdx.x >> 5;
  const int khalf = lane >> 4;
  const int l16 = lane & 15;

  if (threadIdx.x < 64) {
    int idx = tileStart + threadIdx.x;
    if (idx < cnt) { stokS[threadIdx.x] = etok[e * T_TOK + idx];
                     sgateS[threadIdx.x] = egate[e * T_TOK + idx]; }
    else           { stokS[threadIdx.x] = 0; sgateS[threadIdx.x] = 0.f; }
  }
  __syncthreads();

  const _Float16* wsE = wsh + (size_t)e * (2 * IFFN) * DMODEL;
  const _Float16* w2E = w2h + (size_t)e * DMODEL * IFFN;

  // --- Phase A: act[64, I] = silu(X w1^T) * (X v1^T), f16 into LDS ---------
  // jobs: 2 row-pairs x 128 i-col-tiles = 256, round-robin over 8 waves.
  // Row-pair shares B fragments across two A tiles (4 WMMAs per k-step).
  for (int j = wave; j < 2 * (IFFN / 16); j += 8) {
    const int rp = j & 1;                 // rows [rp*32, rp*32+31]
    const int c  = j >> 1;
    const int r0 = rp * 2, r1 = rp * 2 + 1;
    const int tq0 = stokS[r0 * 16 + l16]; // padded rows -> token 0 (safe)
    const int tq1 = stokS[r1 * 16 + l16];
    const _Float16* arow0 = xh + (size_t)(tq0 >> 2) * DMODEL;
    const _Float16* arow1 = xh + (size_t)(tq1 >> 2) * DMODEL;
    const _Float16* w1row = wsE + (size_t)(c * 16 + l16) * DMODEL;
    const _Float16* v1row = wsE + (size_t)(IFFN + c * 16 + l16) * DMODEL;
    // prefetch next col-tile's weight rows into cache ahead of compute
    if (c + 4 < IFFN / 16) {
      __builtin_prefetch(wsE + (size_t)((c + 4) * 16 + l16) * DMODEL, 0, 1);
      __builtin_prefetch(wsE + (size_t)(IFFN + (c + 4) * 16 + l16) * DMODEL, 0, 1);
    }
    v8f accG0 = {}, accU0 = {}, accG1 = {}, accU1 = {};
#pragma unroll 2
    for (int kk = 0; kk < DMODEL / 32; ++kk) {
      const int kb = kk * 32;
      v16h a0 = load_a_f16(arow0, kb, khalf);
      v16h a1 = load_a_f16(arow1, kb, khalf);
      v16h b1 = load_b_f16row(w1row, kb, khalf);
      accG0 = __builtin_amdgcn_wmma_f32_16x16x32_f16(false, a0, false, b1,
                                                     (short)0, accG0, false, false);
      accG1 = __builtin_amdgcn_wmma_f32_16x16x32_f16(false, a1, false, b1,
                                                     (short)0, accG1, false, false);
      v16h b2 = load_b_f16row(v1row, kb, khalf);
      accU0 = __builtin_amdgcn_wmma_f32_16x16x32_f16(false, a0, false, b2,
                                                     (short)0, accU0, false, false);
      accU1 = __builtin_amdgcn_wmma_f32_16x16x32_f16(false, a1, false, b2,
                                                     (short)0, accU1, false, false);
    }
#pragma unroll
    for (int vr = 0; vr < 8; ++vr) {
      const int mm = vr + 8 * khalf;                       // C layout: row in tile
      float act0 = fast_silu(accG0[vr]) * accU0[vr];       // silu(g)*u
      float act1 = fast_silu(accG1[vr]) * accU1[vr];
      actS[(size_t)(r0 * 16 + mm) * IFFN + (c * 16 + l16)] = (_Float16)act0;
      actS[(size_t)(r1 * 16 + mm) * IFFN + (c * 16 + l16)] = (_Float16)act1;
    }
  }
  __syncthreads();

  // --- Phase B: y[64, D] = act @ w2^T, gate-scale, scatter to yslot --------
  // jobs: 2 row-pairs x 64 d-col-tiles = 128; shared B, A from LDS.
  for (int j = wave; j < 2 * (DMODEL / 16); j += 8) {
    const int rp = j & 1;
    const int dc = j >> 1;
    const int r0 = rp * 2, r1 = rp * 2 + 1;
    const _Float16* arow0 = actS + (size_t)(r0 * 16 + l16) * IFFN;  // LDS A rows
    const _Float16* arow1 = actS + (size_t)(r1 * 16 + l16) * IFFN;
    const _Float16* w2row = w2E + (size_t)(dc * 16 + l16) * IFFN;   // contiguous over i
    if (dc + 4 < DMODEL / 16)
      __builtin_prefetch(w2E + (size_t)((dc + 4) * 16 + l16) * IFFN, 0, 1);
    v8f acc0 = {}, acc1 = {};
#pragma unroll 2
    for (int kk = 0; kk < IFFN / 32; ++kk) {
      const int kb = kk * 32;
      v16h a0 = load_a_f16(arow0, kb, khalf);                       // ds_load path
      v16h a1 = load_a_f16(arow1, kb, khalf);
      v16h b  = load_b_f16row(w2row, kb, khalf);
      acc0 = __builtin_amdgcn_wmma_f32_16x16x32_f16(false, a0, false, b,
                                                    (short)0, acc0, false, false);
      acc1 = __builtin_amdgcn_wmma_f32_16x16x32_f16(false, a1, false, b,
                                                    (short)0, acc1, false, false);
    }
#pragma unroll
    for (int vr = 0; vr < 8; ++vr) {
      const int mm = vr + 8 * khalf;
      if (tileStart + r0 * 16 + mm < cnt) {
        const int q = stokS[r0 * 16 + mm];
        const float gate = sgateS[r0 * 16 + mm];
        yslot[(size_t)((q >> 2) * TOPK + (q & 3)) * DMODEL + dc * 16 + l16] =
            gate * acc0[vr];
      }
      if (tileStart + r1 * 16 + mm < cnt) {
        const int q = stokS[r1 * 16 + mm];
        const float gate = sgateS[r1 * 16 + mm];
        yslot[(size_t)((q >> 2) * TOPK + (q & 3)) * DMODEL + dc * 16 + l16] =
            gate * acc1[vr];
      }
    }
  }
}

// ---- Kernel 3: deterministic combine over the 4 slots -----------------------
__global__ __launch_bounds__(256) void moe_combine_kernel(
    const float* __restrict__ yslot, float* __restrict__ out) {
  const int idx = blockIdx.x * 256 + threadIdx.x;   // grid covers T*D exactly
  const int t = idx >> 10;                           // D = 1024
  const int d = idx & (DMODEL - 1);
  const float* p = yslot + (size_t)t * TOPK * DMODEL + d;
  out[idx] = ((p[0] + p[DMODEL]) + (p[2 * DMODEL] + p[3 * DMODEL]));
}

// ---- Host launch ------------------------------------------------------------
extern "C" void kernel_launch(void* const* d_in, const int* in_sizes, int n_in,
                              void* d_out, int out_size, void* d_ws, size_t ws_size,
                              hipStream_t stream) {
  (void)in_sizes; (void)n_in; (void)out_size; (void)ws_size;
  const float* hs  = (const float*)d_in[0];   // [T, D]
  const float* rw  = (const float*)d_in[1];   // [E, D]
  const float* ws  = (const float*)d_in[2];   // [E, 2I, D]
  const float* w2s = (const float*)d_in[3];   // [E, D, I]
  float* out = (float*)d_out;                 // [T, D] fp32

  // workspace layout (~277 MB)
  const size_t NW1 = (size_t)NEXP * 2 * IFFN * DMODEL;   // 67,108,864 elems
  const size_t NW2 = (size_t)NEXP * DMODEL * IFFN;       // 33,554,432 elems
  char* wsp = (char*)d_ws;
  size_t off = 0;
  int*      counts = (int*)(wsp + off);            off += 1024;
  int*      etok   = (int*)(wsp + off);            off += (size_t)NEXP * T_TOK * 4;
  float*    egate  = (float*)(wsp + off);          off += (size_t)NEXP * T_TOK * 4;
  _Float16* xh     = (_Float16*)(wsp + off);       off += (size_t)T_TOK * DMODEL * 2;
  float*    yslot  = (float*)(wsp + off);          off += (size_t)T_TOK * TOPK * DMODEL * 4;
  _Float16* wsh    = (_Float16*)(wsp + off);       off += NW1 * 2;
  _Float16* w2h    = (_Float16*)(wsp + off);       off += NW2 * 2;

  (void)hipMemsetAsync(counts, 0, NEXP * sizeof(int), stream);

  // one-time weight downconversion (read 402 MB, write 201 MB)
  cvt_f32_f16_kernel<<<(int)(NW1 / (8 * 256)), 256, 0, stream>>>(ws, wsh);
  cvt_f32_f16_kernel<<<(int)(NW2 / (8 * 256)), 256, 0, stream>>>(w2s, w2h);

  moe_router_kernel<<<T_TOK / 8, 256, 0, stream>>>(hs, rw, xh, counts, etok, egate);

  const size_t ldsBytes = (size_t)64 * IFFN * 2 + 64 * 4 + 64 * 4;  // 262656 B
  moe_expert_kernel<<<NEXP * (T_TOK / 64), 256, ldsBytes, stream>>>(
      xh, wsh, w2h, counts, etok, egate, yslot);

  moe_combine_kernel<<<(T_TOK * DMODEL) / 256, 256, 0, stream>>>(yslot, out);
}
